// MultiHeadAttention_85615878078828
// MI455X (gfx1250) — compile-verified, matrix operand order checked by reading, not run on hardware
//
#include <hip/hip_runtime.h>

// ---------------------------------------------------------------------------
// MI455X (gfx1250) multi-head attention forward.
// B=2, S=2048, H=1024, N=16 heads, D=64.  All matmuls via
// v_wmma_f32_16x16x32_bf16 (bf16 A/B, f32 accumulate).  Projection GEMMs
// stage weight tiles through LDS with the Tensor Data Mover.
// ---------------------------------------------------------------------------

typedef __attribute__((ext_vector_type(16))) __bf16 v16bf;
typedef __attribute__((ext_vector_type(8)))  float  v8f;
typedef __attribute__((ext_vector_type(4)))  unsigned int v4u;
typedef __attribute__((ext_vector_type(8)))  int    v8i;
typedef __attribute__((ext_vector_type(4)))  int    v4i;

#define WMMA_BF16(A, B, C)                                                    \
  __builtin_amdgcn_wmma_f32_16x16x32_bf16(false, (A), false, (B), (short)0,   \
                                          (C), false, false)

#define B_  2
#define S_  2048
#define H_  1024
#define NH_ 16
#define D_  64
#define M_  (B_ * S_)      // 4096 tokens
#define K3_ (3 * H_)       // 3072

// ---------------------------------------------------------------------------
// TDM: 2D tensor tile load Global -> LDS.  Descriptor bitfields per CDNA5 ISA
// §8.3 (group 0) / §8.4 (group 1).  data_size = 2 bytes (bf16).
// tile_d0 = elements per row, tile_d1 = rows; stride0 = row stride (elems).
// This toolchain exposes the 6-arg builtin:
//   (uint32x4 g0, int32x8 g1, int32x4, int32x4, int32x8, i32 cpol)
// ---------------------------------------------------------------------------
__device__ __forceinline__ void tdm_load_2d_bf16(unsigned lds_off,
                                                 const void* gaddr,
                                                 unsigned tensor_d0,
                                                 unsigned tensor_d1,
                                                 unsigned stride0,
                                                 unsigned tile_d0,
                                                 unsigned tile_d1) {
  unsigned long long ga = (unsigned long long)gaddr;
  v4u g0;
  g0[0] = 1u;                                         // count=1, is_restore=0
  g0[1] = lds_off;                                    // lds_addr (bytes)
  g0[2] = (unsigned)(ga & 0xFFFFFFFFu);               // global_addr[31:0]
  g0[3] = ((unsigned)(ga >> 32) & 0x01FFFFFFu) | (2u << 30);  // [56:32]|type=2
  v8i g1;
  g1[0] = (int)(1u << 16);                            // data_size=1 (2 bytes)
  g1[1] = (int)((tensor_d0 & 0xFFFFu) << 16);         // tensor_dim0[15:0]
  g1[2] = (int)(((tensor_d0 >> 16) & 0xFFFFu) |       // tensor_dim0[31:16]
                ((tensor_d1 & 0xFFFFu) << 16));       // tensor_dim1[15:0]
  g1[3] = (int)(((tensor_d1 >> 16) & 0xFFFFu) |       // tensor_dim1[31:16]
                ((tile_d0 & 0xFFFFu) << 16));         // tile_dim0
  g1[4] = (int)(tile_d1 & 0xFFFFu);                   // tile_dim1 (tile_dim2=0)
  g1[5] = (int)stride0;                               // tensor_dim0_stride lo
  g1[6] = 0;                                          // stride0 hi / stride1 lo
  g1[7] = 0;
  v4i z4 = {0, 0, 0, 0};
  v8i z8 = {0, 0, 0, 0, 0, 0, 0, 0};
  __builtin_amdgcn_tensor_load_to_lds(g0, g1, z4, z4, z8, 0);
}

__device__ __forceinline__ unsigned lds_offset_of(const void* p) {
  // LDS aperture maps the LDS byte offset into addr[31:0].
  return (unsigned)(unsigned long long)p;
}

// Load a 16x32 bf16 WMMA fragment (A-operand pattern; B mirrors it with
// lane = column).  lanes 0-15: K = {k0..k0+7, k0+16..k0+23},
// lanes 16-31: K = {k0+8..k0+15, k0+24..k0+31}.  Two b128 loads.
__device__ __forceinline__ v16bf load_frag(const __bf16* __restrict__ base,
                                           int row, int rowStride, int k0,
                                           int lane) {
  union { uint4 u[2]; v16bf v; } r;
  const int sel = (lane < 16) ? 0 : 8;
  const __bf16* p = base + (size_t)row * (size_t)rowStride + (size_t)(k0 + sel);
  r.u[0] = *(const uint4*)(p);
  r.u[1] = *(const uint4*)(p + 16);
  return r.v;
}

// Same fragment gather out of an LDS tile with 32-element rows (k = 0..31).
__device__ __forceinline__ v16bf lds_frag(const __bf16* base, int row,
                                          int lane) {
  union { uint4 u[2]; v16bf v; } r;
  const int sel = (lane < 16) ? 0 : 8;
  const __bf16* p = base + row * 32 + sel;
  r.u[0] = *(const uint4*)(p);
  r.u[1] = *(const uint4*)(p + 16);
  return r.v;
}

// ---------------------------------------------------------------------------
// f32 -> bf16 bulk convert (vectorized x4)
// ---------------------------------------------------------------------------
__global__ void f32_to_bf16_kernel(const float* __restrict__ src,
                                   __bf16* __restrict__ dst, int n4) {
  int i = blockIdx.x * blockDim.x + threadIdx.x;
  if (i >= n4) return;
  float4 v = ((const float4*)src)[i];
  union { __bf16 h[4]; uint2 u; } pk;
  pk.h[0] = (__bf16)v.x; pk.h[1] = (__bf16)v.y;
  pk.h[2] = (__bf16)v.z; pk.h[3] = (__bf16)v.w;
  ((uint2*)dst)[i] = pk.u;
}

// ---------------------------------------------------------------------------
// GEMM1: QKV = X[4096,1024] * Win[3072,1024]^T + b_in
// Block = 8 waves sharing one 64-feature weight panel; per K-step the
// 64x32 weight tile is TDM-loaded into LDS.  Each wave owns 16 token rows.
// Q,K stored [bh, s, d] bf16; V stored transposed [bh, d, s] bf16.
// ---------------------------------------------------------------------------
__global__ __launch_bounds__(256) void qkv_gemm_kernel(
    const __bf16* __restrict__ X, const __bf16* __restrict__ W,
    const float* __restrict__ bias, __bf16* __restrict__ qbuf,
    __bf16* __restrict__ kbuf, __bf16* __restrict__ vtbuf) {
  __shared__ __bf16 wtile[64 * 32];                  // 4 KB
  const int lane = threadIdx.x & 31;
  const int w    = threadIdx.x >> 5;                 // wave in block: 0..7
  const int nTilesN = K3_ / 64;                      // 48
  const int mb = blockIdx.x / nTilesN;               // 0..31 (128 tokens each)
  const int nt = blockIdx.x % nTilesN;               // 0..47
  const int m0 = mb * 128 + w * 16;
  const int n0 = nt * 64;
  const int lr = lane & 15;

  const unsigned wtile_lds = lds_offset_of(wtile);

  v8f acc[4] = {};
  for (int k0 = 0; k0 < H_; k0 += 32) {
    if (threadIdx.x < 32) {
      // 64 weight rows (features n0..n0+63), 32 k-columns -> LDS [64][32].
      tdm_load_2d_bf16(wtile_lds, W + (size_t)n0 * H_ + (size_t)k0,
                       /*tensor_d0=*/H_, /*tensor_d1=*/K3_,
                       /*stride0=*/H_, /*tile_d0=*/32, /*tile_d1=*/64);
      __builtin_amdgcn_s_wait_tensorcnt(0);
    }
    __syncthreads();
    v16bf a = load_frag(X, m0 + lr, H_, k0, lane);
#pragma unroll
    for (int j = 0; j < 4; ++j) {
      v16bf b = lds_frag(wtile, j * 16 + lr, lane);
      acc[j] = WMMA_BF16(a, b, acc[j]);
    }
    __syncthreads();
  }

  // Epilogue.  C layout: lane = feature column (l%16), VGPR r = token row
  // m0 + r + (lane<16 ? 0 : 8).
  const int which = n0 >> 10;              // 0=Q 1=K 2=V
  const int f0    = n0 & 1023;
  const int head  = f0 >> 6;               // tile is 64-aligned -> one head
  const int b     = m0 >> 11;              // token/2048
  const int s0    = m0 & 2047;
  const int bh    = b * NH_ + head;
  const int rbase = (lane < 16) ? 0 : 8;

#pragma unroll
  for (int j = 0; j < 4; ++j) {
    const int   d  = j * 16 + lr;
    const float bb = bias[n0 + j * 16 + lr];
    if (which == 2) {
      // V transposed: contiguous along s -> one b128 store per c-tile.
      union { __bf16 h[8]; uint4 u; } pk;
#pragma unroll
      for (int r = 0; r < 8; ++r) pk.h[r] = (__bf16)(acc[j][r] + bb);
      *(uint4*)(vtbuf + ((size_t)(bh * D_ + d)) * S_ + (size_t)(s0 + rbase)) =
          pk.u;
    } else {
      __bf16* dst = (which == 0) ? qbuf : kbuf;
#pragma unroll
      for (int r = 0; r < 8; ++r) {
        const int s = s0 + rbase + r;
        dst[((size_t)bh * S_ + (size_t)s) * D_ + (size_t)d] =
            (__bf16)(acc[j][r] + bb);
      }
    }
  }
}

// ---------------------------------------------------------------------------
// Flash attention (causal).  One wave per (bh, 16-query tile).
// S^T = K*Q^T so C-layout chains into the B-operand of O^T = V^T*P^T.
// ---------------------------------------------------------------------------
__global__ __launch_bounds__(128) void attn_kernel(
    const __bf16* __restrict__ qbuf, const __bf16* __restrict__ kbuf,
    const __bf16* __restrict__ vtbuf, __bf16* __restrict__ ctx) {
  const int tid  = blockIdx.x * blockDim.x + threadIdx.x;
  const int wave = tid >> 5;
  const int lane = threadIdx.x & 31;
  const int qt = wave & 127;               // S/16 = 128 query tiles
  const int bh = wave >> 7;                // 0..31
  const int q0 = qt * 16;
  const int lr = lane & 15;
  const int rbase = (lane < 16) ? 0 : 8;

  const __bf16* Q  = qbuf  + (size_t)bh * S_ * D_;
  const __bf16* K  = kbuf  + (size_t)bh * S_ * D_;
  const __bf16* Vt = vtbuf + (size_t)bh * D_ * S_;

  // Q fragments (B-operand): col = q0+lr, K dim = d (64 -> two frags).
  const v16bf qf0 = load_frag(Q, q0 + lr, D_, 0,  lane);
  const v16bf qf1 = load_frag(Q, q0 + lr, D_, 32, lane);

  v8f o[4] = {};
  float mi = -__builtin_inff();
  float li = 0.0f;
  const int   qg    = q0 + lr;             // this lane's query index
  const float scale = 0.125f;              // D^-0.5

  for (int kt0 = 0; kt0 <= q0 + 15; kt0 += 32) {
    // Prefetch the next key tile while we work on this one.
    if (kt0 + 32 <= q0 + 15) {
      __builtin_prefetch(K + (size_t)(kt0 + 32 + lr) * D_, 0, 0);
      __builtin_prefetch(Vt + (size_t)lr * S_ + (size_t)(kt0 + 32), 0, 0);
    }

    // S^T tiles: two key subtiles of 16.
    v8f st[2];
#pragma unroll
    for (int t = 0; t < 2; ++t) {
      v16bf kf0 = load_frag(K, kt0 + t * 16 + lr, D_, 0,  lane);
      v16bf kf1 = load_frag(K, kt0 + t * 16 + lr, D_, 32, lane);
      v8f c = {};
      c = WMMA_BF16(kf0, qf0, c);
      c = WMMA_BF16(kf1, qf1, c);
      st[t] = c;
    }

    // Per-lane: 16 scores for this lane's query column.
    float sv[16];
    float tmax = -__builtin_inff();
#pragma unroll
    for (int t = 0; t < 2; ++t)
#pragma unroll
      for (int r = 0; r < 8; ++r) {
        const int key = kt0 + t * 16 + rbase + r;
        float s = st[t][r] * scale;
        if (key > qg) s = -__builtin_inff();
        sv[t * 8 + r] = s;
        tmax = fmaxf(tmax, s);
      }
    tmax = fmaxf(tmax, __shfl_xor(tmax, 16, 32));

    const float mnew  = fmaxf(mi, tmax);
    const float alpha = __expf(mi - mnew);
    float lsum = 0.0f;
    v16bf pf;
#pragma unroll
    for (int i = 0; i < 16; ++i) {
      const float e = __expf(sv[i] - mnew);
      lsum += e;
      pf[i] = (__bf16)e;                   // pair-packed: exactly B layout
    }
    lsum += __shfl_xor(lsum, 16, 32);
    li = li * alpha + lsum;
    mi = mnew;

#pragma unroll
    for (int j = 0; j < 4; ++j) o[j] = o[j] * alpha;

    // O^T += V^T * P^T   (A = V^T rows d, contiguous in [d,s] storage)
#pragma unroll
    for (int j = 0; j < 4; ++j) {
      v16bf vf = load_frag(Vt, j * 16 + lr, S_, kt0, lane);
      o[j] = WMMA_BF16(vf, pf, o[j]);
    }
  }

  // Normalize + store ctx [token, H] bf16.  Lane holds 8 consecutive d per
  // j-tile for its query -> packed b128 store.
  const float inv  = 1.0f / li;
  const int   b    = bh >> 4;
  const int   head = bh & 15;
  const size_t token = (size_t)b * S_ + (size_t)(q0 + lr);
#pragma unroll
  for (int j = 0; j < 4; ++j) {
    union { __bf16 h[8]; uint4 u; } pk;
#pragma unroll
    for (int r = 0; r < 8; ++r) pk.h[r] = (__bf16)(o[j][r] * inv);
    *(uint4*)(ctx + token * H_ + (size_t)(head * D_ + j * 16 + rbase)) = pk.u;
  }
}

// ---------------------------------------------------------------------------
// GEMM2: out = Ctx[4096,1024] * Wout[1024,1024]^T + b_out   (f32 out)
// Same block-cooperative TDM staging as GEMM1.
// ---------------------------------------------------------------------------
__global__ __launch_bounds__(256) void out_gemm_kernel(
    const __bf16* __restrict__ Ctx, const __bf16* __restrict__ W,
    const float* __restrict__ bias, float* __restrict__ out) {
  __shared__ __bf16 wtile[64 * 32];                  // 4 KB
  const int lane = threadIdx.x & 31;
  const int w    = threadIdx.x >> 5;
  const int nTilesN = H_ / 64;                       // 16
  const int mb = blockIdx.x / nTilesN;               // 0..31
  const int nt = blockIdx.x % nTilesN;               // 0..15
  const int m0 = mb * 128 + w * 16;
  const int n0 = nt * 64;
  const int lr = lane & 15;
  const int rbase = (lane < 16) ? 0 : 8;

  const unsigned wtile_lds = lds_offset_of(wtile);

  v8f acc[4] = {};
  for (int k0 = 0; k0 < H_; k0 += 32) {
    if (threadIdx.x < 32) {
      tdm_load_2d_bf16(wtile_lds, W + (size_t)n0 * H_ + (size_t)k0,
                       /*tensor_d0=*/H_, /*tensor_d1=*/H_,
                       /*stride0=*/H_, /*tile_d0=*/32, /*tile_d1=*/64);
      __builtin_amdgcn_s_wait_tensorcnt(0);
    }
    __syncthreads();
    v16bf a = load_frag(Ctx, m0 + lr, H_, k0, lane);
#pragma unroll
    for (int j = 0; j < 4; ++j) {
      v16bf b = lds_frag(wtile, j * 16 + lr, lane);
      acc[j] = WMMA_BF16(a, b, acc[j]);
    }
    __syncthreads();
  }
#pragma unroll
  for (int j = 0; j < 4; ++j) {
    const int   n  = n0 + j * 16 + lr;
    const float bb = bias[n];
#pragma unroll
    for (int r = 0; r < 8; ++r) {
      const int m = m0 + rbase + r;
      out[(size_t)m * H_ + (size_t)n] = acc[j][r] + bb;
    }
  }
}

// ---------------------------------------------------------------------------
extern "C" void kernel_launch(void* const* d_in, const int* in_sizes, int n_in,
                              void* d_out, int out_size, void* d_ws,
                              size_t ws_size, hipStream_t stream) {
  const float* hidden = (const float*)d_in[0];  // [2,2048,1024]
  const float* w_in   = (const float*)d_in[1];  // [3072,1024]
  const float* b_in   = (const float*)d_in[2];  // [3072]
  const float* w_out  = (const float*)d_in[3];  // [1024,1024]
  const float* b_out  = (const float*)d_in[4];  // [1024]
  float*       out    = (float*)d_out;          // [2,2048,1024]

  char* ws = (char*)d_ws;
  const size_t MB = 1024ull * 1024ull;
  __bf16* Xbf    = (__bf16*)(ws + 0 * MB);   // 4096x1024  (8 MB)
  __bf16* Winbf  = (__bf16*)(ws + 8 * MB);   // 3072x1024  (6 MB)
  __bf16* Woutbf = (__bf16*)(ws + 14 * MB);  // 1024x1024  (2 MB)
  __bf16* Qb     = (__bf16*)(ws + 16 * MB);  // [32,2048,64] (8 MB)
  __bf16* Kb     = (__bf16*)(ws + 24 * MB);  // [32,2048,64] (8 MB)
  __bf16* Vtb    = (__bf16*)(ws + 32 * MB);  // [32,64,2048] (8 MB)
  __bf16* Ctx    = (__bf16*)(ws + 40 * MB);  // 4096x1024  (8 MB)

  // f32 -> bf16 converts (x4 vectorized)
  {
    int n4 = (M_ * H_) / 4;
    f32_to_bf16_kernel<<<n4 / 256, 256, 0, stream>>>(hidden, Xbf, n4);
  }
  {
    int n4 = (K3_ * H_) / 4;
    f32_to_bf16_kernel<<<n4 / 256, 256, 0, stream>>>(w_in, Winbf, n4);
  }
  {
    int n4 = (H_ * H_) / 4;
    f32_to_bf16_kernel<<<n4 / 256, 256, 0, stream>>>(w_out, Woutbf, n4);
  }

  // QKV projection: 32 m-blocks x 48 n-tiles, 8 waves/block
  qkv_gemm_kernel<<<32 * 48, 256, 0, stream>>>(Xbf, Winbf, b_in, Qb, Kb, Vtb);

  // Attention: 32 bh * 128 q-tiles = 4096 waves, 4 waves/block
  attn_kernel<<<4096 / 4, 128, 0, stream>>>(Qb, Kb, Vtb, Ctx);

  // Output projection: 32 m-blocks x 16 n-tiles, 8 waves/block
  out_gemm_kernel<<<32 * 16, 256, 0, stream>>>(Ctx, Woutbf, b_out, out);
}
